// GINEEncoder_3642132267186
// MI455X (gfx1250) — matrix operand order repaired and checked by reading, not compiled, not run on hardware
//
#include <hip/hip_runtime.h>
#include <hip/hip_bf16.h>

#define N_NODES 50000
#define N_EDGES 600000
#define DIM 128
#define EDIM 64

typedef __attribute__((ext_vector_type(16))) __bf16 v16bf;
typedef __attribute__((ext_vector_type(8)))  float  v8f;
typedef __attribute__((ext_vector_type(4)))  int    v4i;

#if defined(__has_builtin)
#if __has_builtin(__builtin_amdgcn_global_load_async_to_lds_b128) && \
    __has_builtin(__builtin_amdgcn_s_wait_asynccnt)
#define USE_ASYNC_LDS 1
#endif
#endif
#ifndef USE_ASYNC_LDS
#define USE_ASYNC_LDS 0
#endif

#if USE_ASYNC_LDS
typedef __attribute__((address_space(1))) v4i* gptr_b128;   // global v4i*
typedef __attribute__((address_space(3))) v4i* lptr_b128;   // LDS v4i*
#endif

__device__ __forceinline__ __bf16 f2bf(float f) {
  unsigned u = __builtin_bit_cast(unsigned, f);
  u += 0x7FFFu + ((u >> 16) & 1u);              // round-to-nearest-even
  unsigned short h = (unsigned short)(u >> 16);
  return __builtin_bit_cast(__bf16, h);
}

// 16-bit A-matrix 16x32 layout (matching B packing): lanes 0-15 hold K 0-15,
// lanes 16-31 hold K 16-31; element e of the v16bf -> K index:
__device__ __forceinline__ int a_k_of(int lane, int e) {
  return ((lane >> 4) << 4) + e;
}

__global__ void k_zero(float* __restrict__ p, long long n) {
  long long i = (long long)blockIdx.x * blockDim.x + threadIdx.x;
  long long stride = (long long)gridDim.x * blockDim.x;
  for (; i < n; i += stride) p[i] = 0.f;
}

// Fused: msg = relu(x[src] + edge_attr @ We + be); atomic scatter-add to aggr[dst].
// Block = 128 threads (4 waves), each wave one 16-edge tile.
__global__ void __launch_bounds__(128)
k_edge(const float* __restrict__ xfeat,         // [N,128] node features
       const long long* __restrict__ eidx,      // [2,E] int64
       const float* __restrict__ ea,            // [E,64]
       const float* __restrict__ We,            // [64,128]
       const float* __restrict__ be,            // [128]
       float* __restrict__ aggr)                // [N,128]
{
  // B fragments for We: [ntile(8)][kchunk(2)][lane(32)] of v16bf  (16 KB)
  __shared__ v16bf sW[8 * 2 * 32];
  __shared__ alignas(16) float sEA[4][16 * EDIM];     // per-wave 16x64 tile (16 KB)

  const int tid = threadIdx.x;
  {
    __bf16* w = (__bf16*)sW;
    for (int i = tid; i < 8 * 2 * 32 * 16; i += 128) {
      int e = i & 15, lane = (i >> 4) & 31, kc = (i >> 9) & 1, t = i >> 10;
      int k = kc * 32 + ((lane & 16) ? 16 : 0) + e;   // B 32x16: lane half selects K half
      int n = t * 16 + (lane & 15);
      w[i] = f2bf(We[k * DIM + n]);
    }
  }
  __syncthreads();

  const int wave = tid >> 5, lane = tid & 31;
  const int tile = blockIdx.x * 4 + wave;
  const long long e0 = (long long)tile * 16;
  if (e0 >= N_EDGES) return;

  if (e0 + 64 < N_EDGES)                       // prefetch next block's tiles
    __builtin_prefetch(ea + (e0 + 64) * EDIM, 0, 1);

  // stage 16x64 f32 edge_attr tile into LDS
  float* sa = sEA[wave];
#if USE_ASYNC_LDS
  {
    const char* g = (const char*)(ea + e0 * EDIM);
    char* s = (char*)sa;
    #pragma unroll
    for (int i = 0; i < 8; ++i) {
      __builtin_amdgcn_global_load_async_to_lds_b128(
          (gptr_b128)(g + (lane + 32 * i) * 16),
          (lptr_b128)(s + (lane + 32 * i) * 16),
          0, 0);
    }
    __builtin_amdgcn_s_wait_asynccnt(0);
  }
#else
  {
    const float4* g = (const float4*)(ea + e0 * EDIM);
    float4* s = (float4*)sa;
    #pragma unroll
    for (int i = 0; i < 8; ++i) s[lane + 32 * i] = g[lane + 32 * i];
  }
#endif

  // src/dst for the 8 output rows this lane covers (D rows m = r + 8*(lane>=16))
  const int moff = (lane & 16) ? 8 : 0;
  long long src[8], dst[8];
  #pragma unroll
  for (int r = 0; r < 8; ++r) {
    src[r] = eidx[e0 + r + moff];
    dst[r] = eidx[(long long)N_EDGES + e0 + r + moff];
  }

  // A fragments (K=0..31 and K=32..63), converted to bf16
  v16bf a0, a1;
  const int m = lane & 15;
  #pragma unroll
  for (int e = 0; e < 16; ++e) {
    int k = a_k_of(lane, e);
    a0[e] = f2bf(sa[m * EDIM + k]);
    a1[e] = f2bf(sa[m * EDIM + 32 + k]);
  }

  #pragma unroll
  for (int t = 0; t < 8; ++t) {
    const v16bf b0 = sW[(t * 2 + 0) * 32 + lane];
    const v16bf b1 = sW[(t * 2 + 1) * 32 + lane];
    v8f acc = {};
    acc = __builtin_amdgcn_wmma_f32_16x16x32_bf16(false, a0, false, b0, (short)0, acc, false, false);
    acc = __builtin_amdgcn_wmma_f32_16x16x32_bf16(false, a1, false, b1, (short)0, acc, false, false);
    const int n = t * 16 + (lane & 15);
    const float bias = be[n];
    #pragma unroll
    for (int r = 0; r < 8; ++r) {
      float v = acc[r] + bias + xfeat[src[r] * DIM + n];
      v = v > 0.f ? v : 0.f;
      atomicAdd(&aggr[dst[r] * DIM + n], v);
    }
  }
}

// out = act((in1 [+ in2]) @ W + bias); optional per-channel sum/sumsq for BN.
// Block = 128 threads (4 waves), each wave one 16-node row tile.
__global__ void __launch_bounds__(128)
k_gemm(const float* __restrict__ in1,
       const float* __restrict__ in2,
       const float* __restrict__ W,             // [128,128]
       const float* __restrict__ bias,          // [128]
       float* __restrict__ out,                 // [N,128]
       float* __restrict__ stats,               // [256] or unused
       int do_add, int do_relu, int do_stats)
{
  __shared__ v16bf sW[8 * 4 * 32];              // B fragments, 32 KB
  const int tid = threadIdx.x;
  {
    __bf16* w = (__bf16*)sW;
    for (int i = tid; i < 8 * 4 * 32 * 16; i += 128) {
      int e = i & 15, lane = (i >> 4) & 31, kc = (i >> 9) & 3, t = i >> 11;
      int k = kc * 32 + ((lane & 16) ? 16 : 0) + e;
      int n = t * 16 + (lane & 15);
      w[i] = f2bf(W[k * DIM + n]);
    }
  }
  __syncthreads();

  const int wave = tid >> 5, lane = tid & 31;
  const int tile = blockIdx.x * 4 + wave;
  const long long r0 = (long long)tile * 16;
  if (r0 >= N_NODES) return;

  const int m = lane & 15;
  const long long rowBase = (r0 + m) * DIM;
  v16bf a[4];
  #pragma unroll
  for (int kc = 0; kc < 4; ++kc) {
    #pragma unroll
    for (int e = 0; e < 16; ++e) {
      int k = kc * 32 + a_k_of(lane, e);
      float v = in1[rowBase + k];
      if (do_add) v += in2[rowBase + k];
      a[kc][e] = f2bf(v);
    }
  }

  const int moff = (lane & 16) ? 8 : 0;
  #pragma unroll
  for (int t = 0; t < 8; ++t) {
    v8f acc = {};
    #pragma unroll
    for (int kc = 0; kc < 4; ++kc) {
      const v16bf b = sW[(t * 4 + kc) * 32 + lane];
      acc = __builtin_amdgcn_wmma_f32_16x16x32_bf16(false, a[kc], false, b, (short)0, acc, false, false);
    }
    const int n = t * 16 + (lane & 15);
    const float bv = bias[n];
    float s1 = 0.f, s2 = 0.f;
    #pragma unroll
    for (int r = 0; r < 8; ++r) {
      float v = acc[r] + bv;
      if (do_relu) v = v > 0.f ? v : 0.f;
      out[(r0 + r + moff) * DIM + n] = v;
      s1 += v; s2 += v * v;
    }
    if (do_stats) {
      atomicAdd(&stats[n], s1);
      atomicAdd(&stats[DIM + n], s2);
    }
  }
}

__global__ void k_bn(float* __restrict__ out, const float* __restrict__ stats,
                     const float* __restrict__ gamma, const float* __restrict__ beta)
{
  const int c = threadIdx.x & (DIM - 1);        // blockDim multiple of 128
  const float invN = 1.0f / (float)N_NODES;
  const float mean = stats[c] * invN;
  float var = stats[DIM + c] * invN - mean * mean;
  const float inv = rsqrtf(var + 1e-5f);
  const float g = gamma[c] * inv;
  const float b = beta[c] - mean * g;
  const long long total = (long long)N_NODES * DIM;
  long long i = (long long)blockIdx.x * blockDim.x + threadIdx.x;
  const long long stride = (long long)gridDim.x * blockDim.x;   // multiple of 128
  for (; i < total; i += stride) out[i] = out[i] * g + b;
}

extern "C" void kernel_launch(void* const* d_in, const int* in_sizes, int n_in,
                              void* d_out, int out_size, void* d_ws, size_t ws_size,
                              hipStream_t stream) {
  const float*     x    = (const float*)d_in[0];
  const long long* ei   = (const long long*)d_in[1];
  const float*     ea   = (const float*)d_in[2];
  const float*     We1  = (const float*)d_in[3];
  const float*     be1  = (const float*)d_in[4];
  const float*     W1   = (const float*)d_in[5];
  const float*     b1   = (const float*)d_in[6];
  const float*     W2   = (const float*)d_in[7];
  const float*     b2   = (const float*)d_in[8];
  const float*     We2  = (const float*)d_in[9];
  const float*     be2  = (const float*)d_in[10];
  const float*     W3   = (const float*)d_in[11];
  const float*     b3   = (const float*)d_in[12];
  const float*     W4   = (const float*)d_in[13];
  const float*     b4   = (const float*)d_in[14];
  const float*     gam  = (const float*)d_in[15];
  const float*     bet  = (const float*)d_in[16];

  float* out   = (float*)d_out;
  float* aggr  = (float*)d_ws;
  float* h     = aggr + (size_t)N_NODES * DIM;
  float* tbuf  = h    + (size_t)N_NODES * DIM;
  float* stats = tbuf + (size_t)N_NODES * DIM;

  const long long nfeat = (long long)N_NODES * DIM;
  dim3 edgeGrid(N_EDGES / 64);                  // 4 tiles of 16 edges per block
  dim3 nodeGrid((N_NODES / 16 + 3) / 4);

  k_zero<<<2048, 256, 0, stream>>>(aggr, nfeat);
  k_zero<<<1, 256, 0, stream>>>(stats, 2 * DIM);

  // conv1: h = relu( relu((x + aggr1) @ W1 + b1) @ W2 + b2 )
  k_edge<<<edgeGrid, 128, 0, stream>>>(x, ei, ea, We1, be1, aggr);
  k_gemm<<<nodeGrid, 128, 0, stream>>>(x, aggr, W1, b1, tbuf, stats, 1, 1, 0);
  k_gemm<<<nodeGrid, 128, 0, stream>>>(tbuf, tbuf, W2, b2, h, stats, 0, 1, 0);

  // conv2: out = relu((h + aggr2) @ W3 + b3) @ W4 + b4   (stats for BN)
  k_zero<<<2048, 256, 0, stream>>>(aggr, nfeat);
  k_edge<<<edgeGrid, 128, 0, stream>>>(h, ei, ea, We2, be2, aggr);
  k_gemm<<<nodeGrid, 128, 0, stream>>>(h, aggr, W3, b3, tbuf, stats, 1, 1, 0);
  k_gemm<<<nodeGrid, 128, 0, stream>>>(tbuf, tbuf, W4, b4, out, stats, 0, 0, 1);

  // batch norm (training-mode batch statistics)
  k_bn<<<1024, 256, 0, stream>>>(out, stats, gam, bet);
}
